// SwiGLU_Bit_88905823027954
// MI455X (gfx1250) — compile-verified, matrix operand order checked by reading, not run on hardware
//
#include <hip/hip_runtime.h>
#include <hip/hip_bf16.h>
#include <cstdint>
#include <cstddef>

#define D_MODEL 2048
#define D_FF    8192
#define M_TOK   8192          // 4 * 2048 tokens
#define GSIZE   64
#define QBF     127.0f
#define EPSQ    1e-5f

typedef __attribute__((ext_vector_type(8))) int v8i;
typedef int b128_t __attribute__((vector_size(16)));   // matches builtin param type

// ---------------------------------------------------------------------------
// CDNA5 async global->LDS copies (ASYNCcnt tracked), with sync fallback.
// ---------------------------------------------------------------------------
#if defined(__has_builtin)
#  if __has_builtin(__builtin_amdgcn_global_load_async_to_lds_b128) && \
      __has_builtin(__builtin_amdgcn_global_load_async_to_lds_b32)
#    define USE_ASYNC 1
#  else
#    define USE_ASYNC 0
#  endif
#else
#  define USE_ASYNC 0
#endif

static __device__ __forceinline__ void cp16_async(void* lds, const void* gsrc) {
#if USE_ASYNC
  __builtin_amdgcn_global_load_async_to_lds_b128(
      (__attribute__((address_space(1))) b128_t*)(uintptr_t)gsrc,
      (__attribute__((address_space(3))) b128_t*)(uintptr_t)lds, 0, 0);
#else
  *(int4*)lds = *(const int4*)gsrc;
#endif
}

static __device__ __forceinline__ void cp4_async(void* lds, const void* gsrc) {
#if USE_ASYNC
  __builtin_amdgcn_global_load_async_to_lds_b32(
      (__attribute__((address_space(1))) int*)(uintptr_t)gsrc,
      (__attribute__((address_space(3))) int*)(uintptr_t)lds, 0, 0);
#else
  *(int*)lds = *(const int*)gsrc;
#endif
}

static __device__ __forceinline__ void async_wait0() {
#if USE_ASYNC
#  if __has_builtin(__builtin_amdgcn_s_wait_asynccnt)
  __builtin_amdgcn_s_wait_asynccnt(0);
#  else
  asm volatile("s_wait_asynccnt 0x0" ::: "memory");
#  endif
#endif
}

static __device__ __forceinline__ v8i frag_zero() {
  v8i z = {0, 0, 0, 0, 0, 0, 0, 0};
  return z;
}

// Load one 16x64 int8 A/B WMMA fragment from an LDS tile stored row-major
// with a 64-byte row pitch (CDNA5 iu8 operand layout) -> 4x ds_load_b64.
static __device__ __forceinline__ v8i load_frag8(const char* lds_tile, int rowBase, int lane) {
  const int r  = rowBase + (lane & 15);
  const int kh = (lane >> 4) << 3;            // 0 or 8
  const char* p = lds_tile + r * 64 + kh;
  int2 d0 = *(const int2*)(p + 0);
  int2 d1 = *(const int2*)(p + 16);
  int2 d2 = *(const int2*)(p + 32);
  int2 d3 = *(const int2*)(p + 48);
  v8i f;
  f[0] = d0.x; f[1] = d0.y; f[2] = d1.x; f[3] = d1.y;
  f[4] = d2.x; f[5] = d2.y; f[6] = d3.x; f[7] = d3.y;
  return f;
}

static __device__ __forceinline__ int quant1(float v, float r) {
  return (int)rintf(fminf(fmaxf(v * r, -QBF), QBF));
}

// ---------------------------------------------------------------------------
// Weight prep: sign(w) -> int8 (+1/0/-1), per-block partial sums of |w|.
// ---------------------------------------------------------------------------
__global__ __launch_bounds__(256) void k_wprep(const float* __restrict__ w,
                                               char* __restrict__ wsgn,
                                               float* __restrict__ partial, int n) {
  __shared__ float red[256];
  float s = 0.f;
  for (int i = blockIdx.x * 256 + threadIdx.x; i < n; i += gridDim.x * 256) {
    float v = w[i];
    wsgn[i] = (v > 0.f) ? (char)1 : ((v < 0.f) ? (char)-1 : (char)0);
    s += fabsf(v);
  }
  red[threadIdx.x] = s;
  __syncthreads();
  for (int o = 128; o > 0; o >>= 1) {
    if (threadIdx.x < o) red[threadIdx.x] += red[threadIdx.x + o];
    __syncthreads();
  }
  if (threadIdx.x == 0) partial[blockIdx.x] = red[0];
}

// Deterministic final reduction: one block per tensor.
__global__ __launch_bounds__(256) void k_wreduce(const float* __restrict__ part,
                                                 float* __restrict__ wsc,
                                                 int nPart, float invN) {
  const float* p = part + (size_t)blockIdx.x * nPart;
  __shared__ float red[256];
  float s = 0.f;
  for (int i = threadIdx.x; i < nPart; i += 256) s += p[i];
  red[threadIdx.x] = s;
  __syncthreads();
  for (int o = 128; o > 0; o >>= 1) {
    if (threadIdx.x < o) red[threadIdx.x] += red[threadIdx.x + o];
    __syncthreads();
  }
  if (threadIdx.x == 0) wsc[blockIdx.x] = red[0] * invN;
}

// ---------------------------------------------------------------------------
// Activation quant: per-group-of-64 symmetric int8 along last dim.
// One wave32 per group: lane holds 2 contiguous floats.
// ---------------------------------------------------------------------------
__global__ __launch_bounds__(256) void k_actq(const float* __restrict__ x,
                                              char* __restrict__ q,
                                              float* __restrict__ sx,
                                              int nGroups, int groupsPerRow) {
  const int lane   = threadIdx.x & 31;
  const int wave   = (blockIdx.x * 256 + threadIdx.x) >> 5;
  const int nWaves = (gridDim.x * 256) >> 5;
  const int rowLen = groupsPerRow * GSIZE;
  for (int g = wave; g < nGroups; g += nWaves) {
    const int row = g / groupsPerRow;
    const int grp = g % groupsPerRow;
    const size_t base = (size_t)row * rowLen + grp * GSIZE + lane * 2;
    float2 v = *(const float2*)(x + base);
    float m = fmaxf(fabsf(v.x), fabsf(v.y));
    for (int o = 16; o > 0; o >>= 1) m = fmaxf(m, __shfl_xor(m, o, 32));
    float s = fmaxf(m, EPSQ);
    float r = QBF / s;
    char2 qq;
    qq.x = (char)quant1(v.x, r);
    qq.y = (char)quant1(v.y, r);
    *(char2*)(q + base) = qq;
    if (lane == 0) sx[(size_t)row * groupsPerRow + grp] = s;
  }
}

// ---------------------------------------------------------------------------
// GEMM1 (fused): y1 = qx @ sign(w1)^T, y2 = qx @ sign(w2)^T,
//                h = silu(y1*ws1) * (y2*ws2), then int8 group-quant of h.
// Block tile 128(M) x 128(N); 8 waves as 2(M) x 4(N); wave tile 64x32
// = 4x2 WMMA 16x16 tiles; K marched in groups of 64 (one iu8 WMMA / group).
// Double-buffered LDS, async global->LDS staging overlapped with WMMA math.
// Per mt-group all 4 WMMAs are issued before any dequant consumes a result,
// keeping 3 independent WMMAs in the IU8 RAW-hazard window (fewer v_nops).
// ---------------------------------------------------------------------------
#define G1_BUFSZ 25088   // A(8192) | B1(8192) | B2(8192) | scales(512)

__global__ __launch_bounds__(256) void k_gemm1(const char* __restrict__ qx,
                                               const float* __restrict__ sx,
                                               const char* __restrict__ w1s,
                                               const char* __restrict__ w2s,
                                               const float* __restrict__ wscale,
                                               char* __restrict__ qh,
                                               float* __restrict__ sh) {
  __shared__ char smem[65536];     // mainloop: 2 x 25088; epilogue: 128x128 f32 h
  float* lH = (float*)smem;

  const int tid   = threadIdx.x;
  const int lane  = tid & 31;
  const int wv    = tid >> 5;
  const int waveM = wv & 1;
  const int waveN = wv >> 1;
  const int blockN = blockIdx.x * 128;
  const int blockM = blockIdx.y * 128;
  const int K  = D_MODEL;
  const int nG = K / GSIZE;                  // 32

  float acc1[4][2][8];
  float acc2[4][2][8];
#pragma unroll
  for (int a = 0; a < 4; ++a)
#pragma unroll
    for (int b = 0; b < 2; ++b)
#pragma unroll
      for (int c = 0; c < 8; ++c) { acc1[a][b][c] = 0.f; acc2[a][b][c] = 0.f; }

  const int lr = tid >> 2;                   // 0..63
  const int lc = (tid & 3) * 16;             // 0,16,32,48

  // stage K-group g into buffer `buf` (async)
  auto stage = [&](int g, int buf) {
    char* base = smem + buf * G1_BUFSZ;
    const size_t kof = (size_t)g * GSIZE + lc;
#pragma unroll
    for (int h = 0; h < 2; ++h) {
      const int r = lr + h * 64;
      cp16_async(base + r * 64 + lc,         qx  + (size_t)(blockM + r) * K + kof);
      cp16_async(base + 8192 + r * 64 + lc,  w1s + (size_t)(blockN + r) * K + kof);
      cp16_async(base + 16384 + r * 64 + lc, w2s + (size_t)(blockN + r) * K + kof);
    }
    if (tid < 128)
      cp4_async(base + 24576 + tid * 4, sx + (size_t)(blockM + tid) * nG + g);
  };

  stage(0, 0);
  for (int g = 0; g < nG; ++g) {
    async_wait0();                 // stage g landed in LDS (this wave's ops)
    __syncthreads();               // visible to all waves
    if (g + 1 < nG) stage(g + 1, (g + 1) & 1);   // overlap with math below

    const char*  base = smem + (g & 1) * G1_BUFSZ;
    const char*  lA   = base;
    const char*  lB1  = base + 8192;
    const char*  lB2  = base + 16384;
    const float* lS   = (const float*)(base + 24576);

    v8i bf1[2], bf2[2];
#pragma unroll
    for (int nt = 0; nt < 2; ++nt) {
      bf1[nt] = load_frag8(lB1, waveN * 32 + nt * 16, lane);
      bf2[nt] = load_frag8(lB2, waveN * 32 + nt * 16, lane);
    }
#pragma unroll
    for (int mt = 0; mt < 4; ++mt) {
      const int mb = waveM * 64 + mt * 16;
      v8i af = load_frag8(lA, mb, lane);
      const float* sp = lS + mb + ((lane >> 4) << 3);
      float4 s0 = *(const float4*)(sp);
      float4 s1 = *(const float4*)(sp + 4);
      float sc[8] = {s0.x, s0.y, s0.z, s0.w, s1.x, s1.y, s1.z, s1.w};
      // issue all 4 independent WMMAs first, then dequantize
      v8i c0 = frag_zero();
      v8i ic[4];
      ic[0] = __builtin_amdgcn_wmma_i32_16x16x64_iu8(true, af, true, bf1[0], c0, false, false);
      ic[1] = __builtin_amdgcn_wmma_i32_16x16x64_iu8(true, af, true, bf2[0], c0, false, false);
      ic[2] = __builtin_amdgcn_wmma_i32_16x16x64_iu8(true, af, true, bf1[1], c0, false, false);
      ic[3] = __builtin_amdgcn_wmma_i32_16x16x64_iu8(true, af, true, bf2[1], c0, false, false);
#pragma unroll
      for (int nt = 0; nt < 2; ++nt) {
#pragma unroll
        for (int i = 0; i < 8; ++i) {
          acc1[mt][nt][i] = fmaf((float)ic[nt * 2 + 0][i], sc[i], acc1[mt][nt][i]);
          acc2[mt][nt][i] = fmaf((float)ic[nt * 2 + 1][i], sc[i], acc2[mt][nt][i]);
        }
      }
    }
    __syncthreads();               // all reads of buf[g&1] done before reuse
  }

  // ---- epilogue: SiLU-gate, stage h tile to LDS --------------------------
  const float ws1 = wscale[0] * (1.0f / 127.0f);
  const float ws2 = wscale[1] * (1.0f / 127.0f);
#pragma unroll
  for (int mt = 0; mt < 4; ++mt) {
#pragma unroll
    for (int nt = 0; nt < 2; ++nt) {
#pragma unroll
      for (int i = 0; i < 8; ++i) {
        const int row = waveM * 64 + mt * 16 + i + ((lane >> 4) << 3);
        const int col = waveN * 32 + nt * 16 + (lane & 15);
        const float y1 = acc1[mt][nt][i] * ws1;
        const float y2 = acc2[mt][nt][i] * ws2;
        lH[row * 128 + col] = (y1 / (1.0f + __expf(-y1))) * y2;
      }
    }
  }
  __syncthreads();

  // ---- group-quantize h: 256 tasks = 128 rows x 2 groups of 64 -----------
  {
    const int row = tid >> 1;
    const int grp = tid & 1;
    const float* hp = lH + row * 128 + grp * 64;
    float m = 0.f;
#pragma unroll 8
    for (int j = 0; j < 64; ++j) m = fmaxf(m, fabsf(hp[j]));
    const float s = fmaxf(m, EPSQ);
    const float r = QBF / s;
    char* qrow = qh + (size_t)(blockM + row) * D_FF + blockN + grp * 64;
#pragma unroll
    for (int c = 0; c < 4; ++c) {            // 4 x 16-byte packed stores
      int d[4];
#pragma unroll
      for (int dw = 0; dw < 4; ++dw) {
        const float* v = hp + c * 16 + dw * 4;
        int b0 = quant1(v[0], r) & 255;
        int b1 = quant1(v[1], r) & 255;
        int b2 = quant1(v[2], r) & 255;
        int b3 = quant1(v[3], r) & 255;
        d[dw] = b0 | (b1 << 8) | (b2 << 16) | (b3 << 24);
      }
      *(int4*)(qrow + c * 16) = make_int4(d[0], d[1], d[2], d[3]);
    }
    sh[(size_t)(blockM + row) * (D_FF / GSIZE) + (blockN / GSIZE) + grp] = s;
  }
}

// ---------------------------------------------------------------------------
// GEMM2: out = (qh @ sign(w3)^T) * ws3/127, fp32 output. M=8192,K=8192,N=2048.
// ---------------------------------------------------------------------------
#define G2_BUFSZ 16896   // A(8192) | B(8192) | scales(512)

__global__ __launch_bounds__(256) void k_gemm2(const char* __restrict__ qh,
                                               const float* __restrict__ sh,
                                               const char* __restrict__ w3s,
                                               const float* __restrict__ wscale3,
                                               float* __restrict__ out) {
  __shared__ char smem[2 * G2_BUFSZ];

  const int tid   = threadIdx.x;
  const int lane  = tid & 31;
  const int wv    = tid >> 5;
  const int waveM = wv & 1;
  const int waveN = wv >> 1;
  const int blockN = blockIdx.x * 128;
  const int blockM = blockIdx.y * 128;
  const int K  = D_FF;
  const int nG = K / GSIZE;                  // 128

  float acc[4][2][8];
#pragma unroll
  for (int a = 0; a < 4; ++a)
#pragma unroll
    for (int b = 0; b < 2; ++b)
#pragma unroll
      for (int c = 0; c < 8; ++c) acc[a][b][c] = 0.f;

  const int lr = tid >> 2;
  const int lc = (tid & 3) * 16;

  auto stage = [&](int g, int buf) {
    char* base = smem + buf * G2_BUFSZ;
    const size_t kof = (size_t)g * GSIZE + lc;
#pragma unroll
    for (int h = 0; h < 2; ++h) {
      const int r = lr + h * 64;
      cp16_async(base + r * 64 + lc,        qh  + (size_t)(blockM + r) * K + kof);
      cp16_async(base + 8192 + r * 64 + lc, w3s + (size_t)(blockN + r) * K + kof);
    }
    if (tid < 128)
      cp4_async(base + 16384 + tid * 4, sh + (size_t)(blockM + tid) * nG + g);
  };

  stage(0, 0);
  for (int g = 0; g < nG; ++g) {
    async_wait0();
    __syncthreads();
    if (g + 1 < nG) stage(g + 1, (g + 1) & 1);

    const char*  base = smem + (g & 1) * G2_BUFSZ;
    const char*  lA   = base;
    const char*  lB   = base + 8192;
    const float* lS   = (const float*)(base + 16384);

    v8i bf[2];
#pragma unroll
    for (int nt = 0; nt < 2; ++nt) bf[nt] = load_frag8(lB, waveN * 32 + nt * 16, lane);
#pragma unroll
    for (int mt = 0; mt < 4; ++mt) {
      const int mb = waveM * 64 + mt * 16;
      v8i af = load_frag8(lA, mb, lane);
      const float* sp = lS + mb + ((lane >> 4) << 3);
      float4 s0 = *(const float4*)(sp);
      float4 s1 = *(const float4*)(sp + 4);
      float sc[8] = {s0.x, s0.y, s0.z, s0.w, s1.x, s1.y, s1.z, s1.w};
      // issue both independent WMMAs, then dequantize
      v8i c0  = frag_zero();
      v8i ic0 = __builtin_amdgcn_wmma_i32_16x16x64_iu8(true, af, true, bf[0], c0, false, false);
      v8i ic1 = __builtin_amdgcn_wmma_i32_16x16x64_iu8(true, af, true, bf[1], c0, false, false);
#pragma unroll
      for (int i = 0; i < 8; ++i) {
        acc[mt][0][i] = fmaf((float)ic0[i], sc[i], acc[mt][0][i]);
        acc[mt][1][i] = fmaf((float)ic1[i], sc[i], acc[mt][1][i]);
      }
    }
    __syncthreads();
  }

  const float ws3 = wscale3[0] * (1.0f / 127.0f);
#pragma unroll
  for (int mt = 0; mt < 4; ++mt) {
#pragma unroll
    for (int nt = 0; nt < 2; ++nt) {
#pragma unroll
      for (int i = 0; i < 8; ++i) {
        const int row = waveM * 64 + mt * 16 + i + ((lane >> 4) << 3);
        const int col = waveN * 32 + nt * 16 + (lane & 15);
        out[(size_t)(blockM + row) * D_MODEL + blockN + col] = acc[mt][nt][i] * ws3;
      }
    }
  }
}

// ---------------------------------------------------------------------------
extern "C" void kernel_launch(void* const* d_in, const int* in_sizes, int n_in,
                              void* d_out, int out_size, void* d_ws, size_t ws_size,
                              hipStream_t stream) {
  const float* x  = (const float*)d_in[0];   // [4,2048,2048]
  const float* w1 = (const float*)d_in[1];   // [8192,2048]
  const float* w2 = (const float*)d_in[2];   // [8192,2048]
  const float* w3 = (const float*)d_in[3];   // [2048,8192]

  char* ws8 = (char*)d_ws;
  size_t off = 0;
  auto alloc = [&](size_t bytes) -> void* {
    void* p = ws8 + off;
    off += (bytes + 255) & ~(size_t)255;
    return p;
  };
  char*  qx   = (char*) alloc((size_t)M_TOK * D_MODEL);             // 16 MB int8
  float* sx   = (float*)alloc((size_t)M_TOK * (D_MODEL/GSIZE) * 4); // 1 MB
  char*  w1s  = (char*) alloc((size_t)D_FF * D_MODEL);              // 16 MB
  char*  w2s  = (char*) alloc((size_t)D_FF * D_MODEL);              // 16 MB
  char*  w3s  = (char*) alloc((size_t)D_MODEL * D_FF);              // 16 MB
  char*  qh   = (char*) alloc((size_t)M_TOK * D_FF);                // 64 MB int8
  float* sh   = (float*)alloc((size_t)M_TOK * (D_FF/GSIZE) * 4);    // 4 MB
  float* part = (float*)alloc((size_t)3 * 4096 * 4);
  float* wsc  = (float*)alloc(256);                                 // [ws1, ws2, ws3]

  const int NW = D_FF * D_MODEL;
  k_wprep<<<4096, 256, 0, stream>>>(w1, w1s, part + 0,    NW);
  k_wprep<<<4096, 256, 0, stream>>>(w2, w2s, part + 4096, NW);
  k_wprep<<<4096, 256, 0, stream>>>(w3, w3s, part + 8192, NW);
  k_wreduce<<<3, 256, 0, stream>>>(part, wsc, 4096, 1.0f / (float)NW);

  k_actq<<<1024, 256, 0, stream>>>(x, qx, sx, M_TOK * (D_MODEL / GSIZE), D_MODEL / GSIZE);

  dim3 g1(D_FF / 128, M_TOK / 128);      // (64, 64)
  k_gemm1<<<g1, 256, 0, stream>>>(qx, sx, w1s, w2s, wsc, qh, sh);

  dim3 g2(D_MODEL / 128, M_TOK / 128);   // (16, 64)
  k_gemm2<<<g2, 256, 0, stream>>>(qh, sh, w3s, wsc + 2, (float*)d_out);
}
